// CKConv_30133490549013
// MI455X (gfx1250) — matrix operand order repaired and checked by reading, not compiled, not run on hardware
//
#include <hip/hip_runtime.h>
#include <math.h>

typedef __attribute__((ext_vector_type(16))) _Float16     v16h;
typedef __attribute__((ext_vector_type(8)))  _Float16     v8h;
typedef __attribute__((ext_vector_type(8)))  float        v8f;
typedef __attribute__((ext_vector_type(4)))  unsigned int v4u;
typedef __attribute__((ext_vector_type(8)))  int          v8i;
typedef __attribute__((ext_vector_type(4)))  int          v4i;

#define OMEGA_0 32.5f

#if defined(__has_builtin)
#if __has_builtin(__builtin_amdgcn_tensor_load_to_lds)
#define CK_HAS_TDM 1
#endif
#endif

// ---------------------------------------------------------------------------
// Kernel 1: SIREN MLP -> h2 [1024 x 32] (f16, row-major, WMMA-A-friendly)
// ---------------------------------------------------------------------------
__global__ void ck_gen_h2(const float* __restrict__ v1, const float* __restrict__ g1,
                          const float* __restrict__ b1, const float* __restrict__ v2,
                          const float* __restrict__ g2, const float* __restrict__ b2,
                          _Float16* __restrict__ h2out) {
  __shared__ float sv2[32 * 32];
  __shared__ float sscale[32];
  int tid = threadIdx.x;
  for (int i = tid; i < 1024; i += blockDim.x) sv2[i] = v2[i];
  __syncthreads();
  if (tid < 32) {
    float s = 0.f;
    #pragma unroll
    for (int jj = 0; jj < 32; ++jj) { float w = sv2[tid * 32 + jj]; s += w * w; }
    sscale[tid] = g2[tid] * rsqrtf(s);   // weight-norm scale: g / ||v_row||
  }
  __syncthreads();
  int j = blockIdx.x * blockDim.x + tid;
  if (j >= 1024) return;
  float t = -1.f + 2.f * (float)j / 1023.f;   // linspace(-1, 1, 1024)
  float h1[32];
  #pragma unroll
  for (int i = 0; i < 32; ++i) {
    float vv = v1[i];
    float w  = g1[i] * vv / fabsf(vv);        // weight_norm of [H,1] rows
    h1[i] = sinf(OMEGA_0 * (t * w + b1[i]));
  }
  #pragma unroll
  for (int i = 0; i < 32; ++i) {
    float acc = 0.f;
    #pragma unroll
    for (int jj = 0; jj < 32; ++jj) acc += sv2[i * 32 + jj] * h1[jj];
    float h2 = sinf(OMEGA_0 * (sscale[i] * acc + b2[i]));
    h2out[j * 32 + i] = (_Float16)h2;
  }
}

// ---------------------------------------------------------------------------
// Kernel 2: krev[d][co][ci] = h2[1023-d] @ W3^T + b3   (WMMA GEMM, 1024x1024x32)
// One wave per 16x16 output tile; 4096 waves.
// ---------------------------------------------------------------------------
__global__ void ck_gen_krev(const _Float16* __restrict__ h2, const float* __restrict__ W3,
                            const float* __restrict__ b3, _Float16* __restrict__ krev) {
  int lane = threadIdx.x & 31;
  int wid  = (blockIdx.x * blockDim.x + threadIdx.x) >> 5;   // 0..4095
  int mt = wid & 63, jt = wid >> 6;
  int jb = jt << 4, mb = mt << 4;
  int koff  = (lane >> 4) * 8;    // A: halves 0..7 -> K=koff..koff+7, 8..15 -> K=16+koff..
  int kbase = (lane >> 4) * 16;   // unused here; B uses koff, same k order as A
  const _Float16* arow = h2 + (size_t)(jb + (lane & 15)) * 32;
  v16h a;
  *(v8h*)&a       = *(const v8h*)(arow + koff);
  *((v8h*)&a + 1) = *(const v8h*)(arow + 16 + koff);
  int m = mb + (lane & 15);
  const float* wrow = W3 + (size_t)m * 32 + koff;
  v16h bmat;
  #pragma unroll
  for (int i = 0; i < 16; ++i) bmat[i] = (_Float16)wrow[i < 8 ? i : i + 8];
  v8f c = {};
  c = __builtin_amdgcn_wmma_f32_16x16x32_f16(false, a, false, bmat,
                                             (short)0, c, false, false);
  float bb = b3[m];
  int rbase = (lane >> 4) * 8;
  #pragma unroll
  for (int r = 0; r < 8; ++r) {
    int jj = jb + r + rbase;                    // output row j of k
    krev[(size_t)(1023 - jj) * 1024 + m] = (_Float16)(c[r] + bb);  // time-reversed
  }
}

// ---------------------------------------------------------------------------
// Kernel 3: zero-padded x in f16: xpad[b][r][ci], r = (l-d)+1024, rows<1024 are 0
// ---------------------------------------------------------------------------
__global__ void ck_pad_x(const float* __restrict__ x, _Float16* __restrict__ xpad) {
  int tid = blockIdx.x * blockDim.x + threadIdx.x;  // < 4*2048*32
  int ci = tid & 31;
  int r  = (tid >> 5) & 2047;
  int b  = tid >> 16;
  float v = 0.f;
  if (r >= 1024) v = x[((size_t)(b << 10) + (r - 1024)) * 32 + ci];
  xpad[tid] = (_Float16)v;
}

// ---------------------------------------------------------------------------
// Kernel 4: causal conv partials.
//  Block = (b, G, c): 8 waves share d-chunk c; wave w handles l-chunk g=8G+w.
//  The 64KB kernel chunk is staged into LDS in two 32KB TDM transfers
//  (TENSOR_LOAD_TO_LDS + s_wait_tensorcnt), shared by all 8 waves.
//  Each tap is a rank-32 update -> one 16x16x32 f16 WMMA per (l-tile, co-half).
// ---------------------------------------------------------------------------
__global__ void ck_conv_partial(const _Float16* __restrict__ xpad,
                                const _Float16* __restrict__ krev,
                                float* __restrict__ partial) {
  __shared__ __align__(128) _Float16 lds_k[16 * 1024];   // 32 KB: 16 taps x [co][ci]
  int c = blockIdx.x & 31;           // d-chunk (32 taps)
  int G = (blockIdx.x >> 5) & 3;     // group of 8 l-chunks
  int b = blockIdx.x >> 7;           // batch
  if (c > 8 * G + 7) return;         // block-uniform triangular skip (before barriers)
  int lane = threadIdx.x & 31;
  int wave = threadIdx.x >> 5;       // 0..7
  int g = 8 * G + wave;              // this wave's l-chunk (32 l positions)
  int koff  = (lane >> 4) * 8;
  int kbase = (lane >> 4) * 16;
  int nrow  = lane & 15;
  v8f acc[2][2] = {};
  const _Float16* xb = xpad + (size_t)b * 2048 * 32;

  for (int s = 0; s < 2; ++s) {
    // ---- stage 16 taps (32 KB of krev) into LDS ----
#if defined(CK_HAS_TDM)
    if (threadIdx.x < 32) {          // wave 0 issues the DMA (TDM ignores EXEC)
      unsigned long long ga =
          (unsigned long long)(uintptr_t)(krev + (size_t)c * 32768 + (size_t)s * 16384);
      unsigned int lbase = (unsigned int)(uintptr_t)(&lds_k[0]);
      v4u d0;                         // D# group 0: count=1, lds_addr, global_addr, type=2
      d0[0] = 1u;
      d0[1] = lbase;
      d0[2] = (unsigned int)ga;
      d0[3] = (unsigned int)(ga >> 32) | (2u << 30);
      v8i d1;                         // D# group 1: data_size=8B, 1-D 4096-unit tile
      d1[0] = (int)(3u << 16);        // workgroup_mask=0, data_size=3 (8B)
      d1[1] = (int)(4096u << 16);     // tensor_dim0[15:0] at bits 63:48
      d1[2] = (int)(1u << 16);        // tensor_dim0[31:16]=0, tensor_dim1=1
      d1[3] = (int)(4096u << 16);     // tile_dim0 = 4096 (32 KB / 8B)
      d1[4] = 1;                      // tile_dim1 = 1
      d1[5] = 4096;                   // tensor_dim0_stride
      d1[6] = 0; d1[7] = 0;
      v4i dz = {0, 0, 0, 0};
#if __has_include(<hip/amd_detail/amd_gfx1250_TDM.h>)
      v8i dz8 = {0, 0, 0, 0, 0, 0, 0, 0};
      __builtin_amdgcn_tensor_load_to_lds(d0, d1, dz, dz, dz8, 0);
#else
      __builtin_amdgcn_tensor_load_to_lds(d0, d1, dz, dz, 0);
#endif
      __builtin_amdgcn_s_wait_tensorcnt(0);
    }
    __syncthreads();
#else
    {                                 // fallback: cooperative vector copy
      const v8h* src = (const v8h*)(krev + (size_t)c * 32768 + (size_t)s * 16384);
      v8h* dst = (v8h*)lds_k;
      for (int i = threadIdx.x; i < 2048; i += 256) dst[i] = src[i];
    }
    __syncthreads();
#endif
    // ---- consume 16 taps from LDS ----
    for (int dd = 0; dd < 16; ++dd) {
      int d = (c << 5) + (s << 4) + dd;
      const _Float16* brow = lds_k + dd * 1024;
      v16h bf[2];
      #pragma unroll
      for (int nh = 0; nh < 2; ++nh)
        bf[nh] = *(const v16h*)(brow + ((nh << 4) + nrow) * 32 + kbase);  // ds_load
      #pragma unroll
      for (int tt = 0; tt < 2; ++tt) {
        int l  = (g << 5) + (tt << 4) + nrow;
        int rr = l - d + 1024;         // >= 769; rows < 1024 are zeros (causal pad)
        const _Float16* arow = xb + (size_t)rr * 32;
        v16h a;
        a = *(const v16h*)(arow + kbase);
        #pragma unroll
        for (int nh = 0; nh < 2; ++nh)
          acc[tt][nh] = __builtin_amdgcn_wmma_f32_16x16x32_f16(
              false, a, false, bf[nh], (short)0, acc[tt][nh], false, false);
      }
    }
    __syncthreads();                   // all waves done with LDS before restaging
  }

  float* P = partial + (size_t)(c * 4 + b) * 1024 * 32;
  int rbase = (lane >> 4) * 8;
  #pragma unroll
  for (int tt = 0; tt < 2; ++tt)
    #pragma unroll
    for (int nh = 0; nh < 2; ++nh) {
      int co = (nh << 4) + nrow;
      #pragma unroll
      for (int r = 0; r < 8; ++r) {
        int l = (g << 5) + (tt << 4) + r + rbase;
        P[(size_t)l * 32 + co] = acc[tt][nh][r];
      }
    }
}

// ---------------------------------------------------------------------------
// Kernel 5: deterministic reduction over d-chunks + bias. A block (b,G,c) was
// active iff c <= 8G+7, so slot (c,l) is written iff c <= 8*(l>>8)+7.
// ---------------------------------------------------------------------------
__global__ void ck_reduce(const float* __restrict__ partial, const float* __restrict__ bias,
                          float* __restrict__ out) {
  int tid = blockIdx.x * blockDim.x + threadIdx.x;  // < 4*1024*32
  int co = tid & 31;
  int l  = (tid >> 5) & 1023;
  int b  = tid >> 15;
  int cmax = ((l >> 8) << 3) + 7;
  if (cmax > 31) cmax = 31;
  float s = bias[co];
  const float* P = partial + ((size_t)b * 1024 + l) * 32 + co;
  for (int cc = 0; cc <= cmax; ++cc) s += P[(size_t)cc * 131072];
  out[tid] = s;   // [B, L, Cout] row-major
}

extern "C" void kernel_launch(void* const* d_in, const int* in_sizes, int n_in,
                              void* d_out, int out_size, void* d_ws, size_t ws_size,
                              hipStream_t stream) {
  (void)in_sizes; (void)n_in; (void)out_size; (void)ws_size;
  const float* x    = (const float*)d_in[0];
  const float* v1   = (const float*)d_in[1];
  const float* g1   = (const float*)d_in[2];
  const float* b1   = (const float*)d_in[3];
  const float* v2   = (const float*)d_in[4];
  const float* g2   = (const float*)d_in[5];
  const float* b2   = (const float*)d_in[6];
  const float* W3   = (const float*)d_in[7];
  const float* b3   = (const float*)d_in[8];
  const float* bias = (const float*)d_in[9];
  float* out = (float*)d_out;

  char* ws = (char*)d_ws;
  _Float16* h2f16 = (_Float16*)(ws);                 //   64 KB: h2 [1024][32] f16
  _Float16* krev  = (_Float16*)(ws + 65536);         //    2 MB: krev [1024][32][32] f16
  _Float16* xpad  = (_Float16*)(ws + 2162688);       //  512 KB: xpad [4][2048][32] f16
  float*    part  = (float*)   (ws + 2686976);       //   16 MB: partial [32][4][1024][32] f32

  ck_gen_h2<<<4, 256, 0, stream>>>(v1, g1, b1, v2, g2, b2, h2f16);
  ck_pad_x<<<1024, 256, 0, stream>>>(x, xpad);
  ck_gen_krev<<<512, 256, 0, stream>>>(h2f16, W3, b3, krev);   // 4096 waves, WMMA GEMM
  ck_conv_partial<<<512, 256, 0, stream>>>(xpad, krev, part);  // TDM->LDS + WMMA conv
  ck_reduce<<<512, 256, 0, stream>>>(part, bias, out);
}
